// WaveFunctionOnT_50861002719648
// MI455X (gfx1250) — matrix operand order repaired and checked by reading, not compiled, not run on hardware
//
#include <hip/hip_runtime.h>
#include <math.h>

typedef __attribute__((ext_vector_type(16))) __bf16 v16bf;
typedef __attribute__((ext_vector_type(8)))  __bf16 v8bf;
typedef __attribute__((ext_vector_type(8)))  float  v8f;

#define NSTREAM 5
#define NCOMBO  31
#define DDIM    64
#define MBLOCK  128
#define BS      (8 * 2048)      // B*S rows
#define KMAX    (5 * DDIM)      // 320
#define PAD     8               // bf16 elements of row padding (16B) -> conflict-free ds_read_b128

struct WfParams {
    const float* streams[NSTREAM];
    const float* Wa[NCOMBO];
    const float* ba[NCOMBO];
    const float* Wp[NCOMBO];
    const float* bp[NCOMBO];
    float pos[NCOMBO];
    int   cstream[NCOMBO][NSTREAM];
    int   ck[NCOMBO];
    float* out;
};

__global__ __launch_bounds__(256) void wavefn_wmma_kernel(WfParams P) {
    extern __shared__ __bf16 smem[];
    const int ci     = blockIdx.y;
    const int kdepth = P.ck[ci];
    const int K      = kdepth * DDIM;          // fan_in, multiple of 64
    const int stride = K + PAD;                // bf16 elements per LDS row
    __bf16* Xs = smem;                         // [MBLOCK][stride]   rows m, cols K
    __bf16* Ws = smem + MBLOCK * stride;       // [128][stride]      rows n (0..63 Wa | 64..127 Wp), cols K

    const int tid     = threadIdx.x;
    const int rowBase = blockIdx.x * MBLOCK;

    // ---- Stage X tile (concat of selected streams) into LDS as bf16 ----
    for (int t = 0; t < kdepth; ++t) {
        const float* sp = P.streams[P.cstream[ci][t]];
        for (int idx = tid; idx < MBLOCK * DDIM; idx += 256) {
            int row = idx >> 6, d = idx & 63;
            float v = sp[(size_t)(rowBase + row) * DDIM + d];
            Xs[row * stride + t * DDIM + d] = (__bf16)v;
        }
    }
    // ---- Stage weights TRANSPOSED [n][K] into LDS as bf16 ----
    {
        const float* wa = P.Wa[ci];
        const float* wp = P.Wp[ci];
        const int total = K * DDIM;
        for (int idx = tid; idx < 2 * total; idx += 256) {
            int sel = idx >= total;            // 0 = Wa, 1 = Wp
            int j   = sel ? idx - total : idx;
            int kk  = j >> 6, n = j & 63;      // global W is [K][64] row-major
            float v = (sel ? wp : wa)[j];
            Ws[(n + (sel << 6)) * stride + kk] = (__bf16)v;
        }
    }
    __syncthreads();

    const int wave = tid >> 5;
    const int lane = tid & 31;
    const int half = lane >> 4;   // K-half selector per ISA fragment layouts
    const int l15  = lane & 15;

    v8f acc[8];
    const v8f vzero = {0.f, 0.f, 0.f, 0.f, 0.f, 0.f, 0.f, 0.f};
#pragma unroll
    for (int f = 0; f < 8; ++f) acc[f] = vzero;

    // A fragment source row for this lane (M = l15 within the wave's 16-row slice)
    const __bf16* arow = Xs + (wave * 16 + l15) * stride;

    for (int kb = 0; kb < K; kb += 32) {
        // 16-bit A 16x32 layout: lane(half,m); elems 0..7 -> K = kb + half*8 + e,
        //                                     elems 8..15 -> K = kb + 16 + half*8 + (e-8)
        v8bf a_lo = *(const v8bf*)(arow + kb + half * 8);
        v8bf a_hi = *(const v8bf*)(arow + kb + 16 + half * 8);
        v16bf a = __builtin_shufflevector(a_lo, a_hi,
                                          0, 1, 2, 3, 4, 5, 6, 7, 8, 9, 10, 11, 12, 13, 14, 15);
#pragma unroll
        for (int f = 0; f < 8; ++f) {
            // 16-bit B 32x16 layout: lane(half,n); elems 0..7 -> K = kb + half*8 + e, elems 8..15 -> K = kb + 16 + half*8 + (e-8), N = l15
            const __bf16* brow = Ws + (f * 16 + l15) * stride + kb + half * 8;
            v8bf b_lo = *(const v8bf*)(brow);
            v8bf b_hi = *(const v8bf*)(brow + 16);
            v16bf b = __builtin_shufflevector(b_lo, b_hi,
                                              0, 1, 2, 3, 4, 5, 6, 7, 8, 9, 10, 11, 12, 13, 14, 15);
            acc[f] = __builtin_amdgcn_wmma_f32_16x16x32_bf16(
                false, a, false, b, (short)0, acc[f], false, false);
        }
    }

    // ---- Fused epilogue: softplus / phase / psi = amp * e^{i*phase} ----
    const float* ba  = P.ba[ci];
    const float* bp  = P.bp[ci];
    const float  pos = P.pos[ci];
    float* out = P.out;
    const size_t T = (size_t)BS * NCOMBO * DDIM;   // element count of one [B,S,N,D] tensor

#pragma unroll
    for (int f = 0; f < 4; ++f) {
        const int   n   = f * 16 + l15;            // output feature d in 0..63
        const float ban = ba[n];
        const float bpn = bp[n] + pos;
#pragma unroll
        for (int i = 0; i < 8; ++i) {
            // C/D layout: VGPR i -> M = i (lanes 0-15) or 8+i (lanes 16-31), N = l15
            const int    m   = i + half * 8;
            const size_t r   = (size_t)(rowBase + wave * 16 + m);
            const size_t idx = (r * NCOMBO + ci) * DDIM + n;

            float apre = acc[f][i] + ban;          // amp pre-activation (cols 0..63)
            float ph   = acc[f + 4][i] + bpn;      // phase (cols 64..127)

            float amp;
            if (apre > 20.f)        amp = apre;            // softplus(x) ~ x
            else if (apre < -20.f)  amp = __expf(apre);    // softplus(x) ~ e^x
            else                    amp = __logf(1.f + __expf(apre));

            float sn = __sinf(ph), cs = __cosf(ph);
            float2 psi = make_float2(amp * cs, amp * sn);
            *(float2*)(out + 2 * idx) = psi;               // complex64 interleaved
            out[2 * T + idx] = amp;
            out[3 * T + idx] = ph;
        }
    }
}

extern "C" void kernel_launch(void* const* d_in, const int* in_sizes, int n_in,
                              void* d_out, int out_size, void* d_ws, size_t ws_size,
                              hipStream_t stream) {
    (void)in_sizes; (void)d_ws; (void)ws_size; (void)out_size;

    WfParams prm;
    for (int i = 0; i < NSTREAM; ++i) prm.streams[i] = (const float*)d_in[i];
    prm.out = (float*)d_out;

    // Combos in torch/itertools order: depth k = 1..5, lexicographic within depth.
    const double PI = 3.14159265358979323846;
    double ang[NSTREAM];
    for (int i = 0; i < NSTREAM; ++i) ang[i] = 2.0 * PI * (double)i / (1.0 + sqrt(5.0)) * 2.0;

    int nc = 0;
    for (int kk = 1; kk <= NSTREAM; ++kk) {
        int idx[NSTREAM];
        for (int i = 0; i < kk; ++i) idx[i] = i;
        for (;;) {
            prm.ck[nc] = kk;
            double s = 0.0;
            for (int i = 0; i < NSTREAM; ++i) prm.cstream[nc][i] = 0;
            for (int i = 0; i < kk; ++i) { prm.cstream[nc][i] = idx[i]; s += ang[idx[i]]; }
            prm.pos[nc] = (float)s;
            ++nc;
            int i = kk - 1;
            while (i >= 0 && idx[i] == NSTREAM - kk + i) --i;
            if (i < 0) break;
            ++idx[i];
            for (int j = i + 1; j < kk; ++j) idx[j] = idx[j - 1] + 1;
        }
    }
    // nc == 31

    if (n_in >= 5 + 4 * NCOMBO) {
        // Lists flattened elementwise: streams(5), Wa(31), ba(31), Wp(31), bp(31)
        for (int i = 0; i < NCOMBO; ++i) {
            prm.Wa[i] = (const float*)d_in[5 + i];
            prm.ba[i] = (const float*)d_in[5 + NCOMBO + i];
            prm.Wp[i] = (const float*)d_in[5 + 2 * NCOMBO + i];
            prm.bp[i] = (const float*)d_in[5 + 3 * NCOMBO + i];
        }
    } else {
        // Fallback: each list concatenated into one flat buffer (9 inputs)
        const float* waB = (const float*)d_in[5];
        const float* baB = (const float*)d_in[6];
        const float* wpB = (const float*)d_in[7];
        const float* bpB = (const float*)d_in[8];
        size_t woff = 0;
        for (int i = 0; i < NCOMBO; ++i) {
            prm.Wa[i] = waB + woff;
            prm.Wp[i] = wpB + woff;
            prm.ba[i] = baB + (size_t)i * DDIM;
            prm.bp[i] = bpB + (size_t)i * DDIM;
            woff += (size_t)prm.ck[i] * DDIM * DDIM;
        }
    }

    const size_t shmem = (size_t)(MBLOCK + 2 * DDIM) * (KMAX + PAD) * sizeof(__bf16); // ~164 KB
    (void)hipFuncSetAttribute(reinterpret_cast<const void*>(wavefn_wmma_kernel),
                              hipFuncAttributeMaxDynamicSharedMemorySize, (int)shmem);

    dim3 grid(BS / MBLOCK, NCOMBO, 1);   // 128 row-tiles x 31 combos
    dim3 block(256, 1, 1);               // 8 waves (wave32)
    wavefn_wmma_kernel<<<grid, block, shmem, stream>>>(prm);
}
